// Attention_74474732913237
// MI455X (gfx1250) — matrix operand order repaired and checked by reading, not compiled, not run on hardware
//
#include <hip/hip_runtime.h>

// ============================================================================
// MI455X (gfx1250) fused cross-attention:
//   q = relu(x@wq+bq); k = relu(y@wk+bk)@wg; v = relu(y@wv+bv)
//   w = softmax(mask(q@k^T/sqrt(A))); attention = w@v
// All GEMMs are NT (B stored N-major) -> every LDS fragment read is b128.
// V_WMMA_F32_16X16X32_F16 everywhere; async global->LDS tile staging
// (GLOBAL_LOAD_ASYNC_TO_LDS_B128 + s_wait_asynccnt) when available.
// Workspace requirement: 328 MiB (overlayed buffers, stream-ordered reuse).
// ============================================================================

typedef __attribute__((ext_vector_type(16))) _Float16 v16h;
typedef __attribute__((ext_vector_type(8)))  _Float16 v8h;
typedef __attribute__((ext_vector_type(4)))  _Float16 v4h;
typedef __attribute__((ext_vector_type(8)))  float    v8f;

#define DEV __device__ __forceinline__

static constexpr int BM = 128;   // block tile M
static constexpr int BN = 128;   // block tile N
static constexpr int BK = 32;    // K per WMMA f16 instruction

#if __has_builtin(__builtin_amdgcn_global_load_async_to_lds_b128) && \
    __has_builtin(__builtin_amdgcn_s_wait_asynccnt)
#define USE_ASYNC 1
#else
#define USE_ASYNC 0
#endif

#if USE_ASYNC
// Builtin signature (from hipcc diagnostic): params are pointers to
// int __attribute__((vector_size(16))) in AS1 (global) / AS3 (LDS).
typedef int b128_t __attribute__((vector_size(16)));
typedef __attribute__((address_space(1))) b128_t* glb_b128_p;
typedef __attribute__((address_space(3))) b128_t* lds_b128_p;

// 16-byte async copy, global -> LDS, tracked by ASYNCcnt (no VGPR staging).
DEV void async_cp16(const _Float16* g, _Float16* l) {
  __builtin_amdgcn_global_load_async_to_lds_b128((glb_b128_p)g, (lds_b128_p)l,
                                                 0, 0);
}
#endif

DEV v16h cat8(v8h lo, v8h hi) {
  v16h r;
#pragma unroll
  for (int i = 0; i < 8; ++i) { r[i] = lo[i]; r[i + 8] = hi[i]; }
  return r;
}

// A fragment (16x32 f16) from As[BM][BK]. ISA layout: lane L<16 = row M=L,
// K {0..7,16..23}; lane L+16 = row M=L, K {8..15,24..31}.
DEV v16h frag_a(const _Float16* As, int mbase, int lane) {
  const int r  = mbase + (lane & 15);
  const int ko = (lane >> 4) * 8;              // 0 or 8
  v8h lo = *(const v8h*)(As + r * BK + ko);
  v8h hi = *(const v8h*)(As + r * BK + ko + 16);
  return cat8(lo, hi);
}

// B fragment for NT GEMM from Bs[BN][BK] (rows = N, contiguous K):
// two b128 LDS reads per fragment.
DEV v16h frag_bt(const _Float16* Bs, int nbase, int lane) {
  const int c  = nbase + (lane & 15);
  const int k0 = (lane >> 4) * 8;              // 0 or 8
  v8h lo = *(const v8h*)(Bs + c * BK + k0);
  v8h hi = *(const v8h*)(Bs + c * BK + k0 + 16);
  return cat8(lo, hi);
}

DEV v8f wmma32(v16h a, v16h b, v8f c) {
  // (neg_a, A, neg_b, B, c_mod, C, reuse_a, reuse_b)
  return __builtin_amdgcn_wmma_f32_16x16x32_f16(false, a, false, b, (short)0, c,
                                                false, false);
}

// ---------------------------------------------------------------------------
// f32 -> f16, 4 elems/thread (b128 load, b64 store)
// ---------------------------------------------------------------------------
__global__ __launch_bounds__(256)
void cvt_f32_f16(const float* __restrict__ src, _Float16* __restrict__ dst,
                 long long n) {
  long long i = ((long long)blockIdx.x * blockDim.x + threadIdx.x) * 4;
  if (i + 3 < n) {
    float4 f = *(const float4*)(src + i);
    v4h h = {(_Float16)f.x, (_Float16)f.y, (_Float16)f.z, (_Float16)f.w};
    *(v4h*)(dst + i) = h;
  }
}

// ---------------------------------------------------------------------------
// f32 [R,C] -> f16 transposed [C,R]  (32x32 LDS tiles; for weight matrices)
// ---------------------------------------------------------------------------
__global__ __launch_bounds__(256)
void cvt_transpose_f32_f16(const float* __restrict__ src,
                           _Float16* __restrict__ dst, int R, int C) {
  __shared__ _Float16 t[32][33];
  const int bx = blockIdx.x * 32;   // source column block
  const int by = blockIdx.y * 32;   // source row block
  const int tx = threadIdx.x & 31, ty = threadIdx.x >> 5;  // 32 x 8
#pragma unroll
  for (int i = 0; i < 32; i += 8)
    t[ty + i][tx] = (_Float16)src[(size_t)(by + ty + i) * C + bx + tx];
  __syncthreads();
#pragma unroll
  for (int i = 0; i < 32; i += 8)
    dst[(size_t)(bx + ty + i) * R + by + tx] = t[tx][ty + i];
}

// ---------------------------------------------------------------------------
// Unified batched NT GEMM:  C = epilogue(A[M,K] @ Bt[N,K]^T)
//   OUT: 0 = f16 row-major, 1 = f32 row-major, 2 = f16 transposed (col-major)
//   RELU/BIAS: projection epilogue.  MASK: scores epilogue (scale + -1e30).
// 256 threads = 8 waves (2x4); wave does 64x32 = 4x2 WMMA tiles.
// Double-buffered LDS; async global->LDS staging overlapped with WMMA.
// ---------------------------------------------------------------------------
template <int OUT, int RELU, int BIAS, int MASK>
__global__ __launch_bounds__(256, 2)
void gemm_nt(const _Float16* __restrict__ A, const _Float16* __restrict__ Bt,
             const float* __restrict__ bias,
             const unsigned char* __restrict__ mask, void* __restrict__ Cout,
             int M, int N, int K, float scale,
             unsigned sA, unsigned sB, unsigned sC, unsigned sM) {
  __shared__ __align__(16) _Float16 As[2][BM * BK];   // 8 KB x2
  __shared__ __align__(16) _Float16 Bs[2][BN * BK];   // 8 KB x2

  A  += (size_t)blockIdx.z * sA;
  Bt += (size_t)blockIdx.z * sB;
  if (MASK) mask += (size_t)blockIdx.z * sM;

  const int tid = threadIdx.x, lane = tid & 31, wave = tid >> 5;
  const int wm = (wave >> 2) * 64, wn = (wave & 3) * 32;
  const int m0 = blockIdx.y * BM, n0 = blockIdx.x * BN;

  // staging: 256 threads cover a 128-row x 32-col f16 tile in 16-half chunks
  const int rr = tid >> 1, cc = (tid & 1) * 16;
  const int lo = rr * BK + cc;
  const _Float16* aRow = A  + (size_t)(m0 + rr) * K + cc;
  const _Float16* bRow = Bt + (size_t)(n0 + rr) * K + cc;

  v8f acc[4][2] = {};
  const int KT = K / BK;

#if USE_ASYNC
  // ------- async pipeline: copies for buffer nxt run under the WMMAs -------
  async_cp16(aRow, &As[0][lo]);
  async_cp16(aRow + 8, &As[0][lo + 8]);
  async_cp16(bRow, &Bs[0][lo]);
  async_cp16(bRow + 8, &Bs[0][lo + 8]);
  __builtin_amdgcn_s_wait_asynccnt(0);
  __syncthreads();

#pragma unroll 2
  for (int kt = 0; kt < KT; ++kt) {
    const int cur = kt & 1, nxt = cur ^ 1;
    if (kt + 1 < KT) {
      const int k1 = (kt + 1) * BK;
      async_cp16(aRow + k1, &As[nxt][lo]);
      async_cp16(aRow + k1 + 8, &As[nxt][lo + 8]);
      async_cp16(bRow + k1, &Bs[nxt][lo]);
      async_cp16(bRow + k1 + 8, &Bs[nxt][lo + 8]);
      if (kt + 2 < KT) {                      // warm L2 for tile kt+2
        __builtin_prefetch(aRow + k1 + BK, 0, 1);
        __builtin_prefetch(bRow + k1 + BK, 0, 1);
      }
    }

    v16h af[4], bf[2];
#pragma unroll
    for (int tm = 0; tm < 4; ++tm) af[tm] = frag_a(As[cur], wm + tm * 16, lane);
#pragma unroll
    for (int tn = 0; tn < 2; ++tn) bf[tn] = frag_bt(Bs[cur], wn + tn * 16, lane);
#pragma unroll
    for (int tm = 0; tm < 4; ++tm)
#pragma unroll
      for (int tn = 0; tn < 2; ++tn)
        acc[tm][tn] = wmma32(af[tm], bf[tn], acc[tm][tn]);

    if (kt + 1 < KT) __builtin_amdgcn_s_wait_asynccnt(0);
    __syncthreads();
  }
#else
  // ------------- fallback: global->VGPR->LDS double buffering -------------
  uint4 a0, a1, b0, b1;
  a0 = *(const uint4*)(aRow);     a1 = *(const uint4*)(aRow + 8);
  b0 = *(const uint4*)(bRow);     b1 = *(const uint4*)(bRow + 8);
  *(uint4*)&As[0][lo] = a0;       *(uint4*)&As[0][lo + 8] = a1;
  *(uint4*)&Bs[0][lo] = b0;       *(uint4*)&Bs[0][lo + 8] = b1;
  __syncthreads();

#pragma unroll 2
  for (int kt = 0; kt < KT; ++kt) {
    const int cur = kt & 1, nxt = cur ^ 1;
    if (kt + 1 < KT) {
      const int k1 = (kt + 1) * BK;
      a0 = *(const uint4*)(aRow + k1);  a1 = *(const uint4*)(aRow + k1 + 8);
      b0 = *(const uint4*)(bRow + k1);  b1 = *(const uint4*)(bRow + k1 + 8);
      if (kt + 2 < KT) {
        __builtin_prefetch(aRow + k1 + BK, 0, 1);
        __builtin_prefetch(bRow + k1 + BK, 0, 1);
      }
    }
    v16h af[4], bf[2];
#pragma unroll
    for (int tm = 0; tm < 4; ++tm) af[tm] = frag_a(As[cur], wm + tm * 16, lane);
#pragma unroll
    for (int tn = 0; tn < 2; ++tn) bf[tn] = frag_bt(Bs[cur], wn + tn * 16, lane);
#pragma unroll
    for (int tm = 0; tm < 4; ++tm)
#pragma unroll
      for (int tn = 0; tn < 2; ++tn)
        acc[tm][tn] = wmma32(af[tm], bf[tn], acc[tm][tn]);
    __syncthreads();
    if (kt + 1 < KT) {
      *(uint4*)&As[nxt][lo] = a0;  *(uint4*)&As[nxt][lo + 8] = a1;
      *(uint4*)&Bs[nxt][lo] = b0;  *(uint4*)&Bs[nxt][lo + 8] = b1;
    }
    __syncthreads();
  }
#endif

  // Epilogue. 16x16 f32 D layout: lanes 0-15 -> M=r, lanes 16-31 -> M=r+8.
  // All output offsets fit in u32 (largest tensor = 100.7M elems).
  const int cn = lane & 15, radd = (lane >> 4) * 8;
  const unsigned cbase = (unsigned)blockIdx.z * sC;
#pragma unroll
  for (int tm = 0; tm < 4; ++tm) {
#pragma unroll
    for (int tn = 0; tn < 2; ++tn) {
      const int ncol = n0 + wn + tn * 16 + cn;
      const float bb = BIAS ? bias[ncol] : 0.0f;
      const bool keep = MASK ? (mask[ncol] != 0) : true;
#pragma unroll
      for (int r = 0; r < 8; ++r) {
        const int mrow = m0 + wm + tm * 16 + radd + r;
        float v = acc[tm][tn][r];
        if (MASK) { v *= scale; if (!keep) v = -1e30f; }
        if (BIAS) v += bb;
        if (RELU) v = fmaxf(v, 0.0f);
        unsigned idx;
        if (OUT == 2) idx = cbase + (unsigned)ncol * (unsigned)M + (unsigned)mrow;
        else          idx = cbase + (unsigned)mrow * (unsigned)N + (unsigned)ncol;
        if (OUT == 1) ((float*)Cout)[idx] = v;
        else          ((_Float16*)Cout)[idx] = (_Float16)v;
      }
    }
  }
}

// ---------------------------------------------------------------------------
// Row softmax over LY=2048: one 256-thread block (8 waves) per row.
// wave32 shfl_xor reduce + 8-slot LDS combine. Writes normalized f32 in place
// and f16 probabilities for the P@v WMMA GEMM.
// ---------------------------------------------------------------------------
__global__ __launch_bounds__(256)
void softmax_rows(float* __restrict__ W, _Float16* __restrict__ P16, int LY) {
  constexpr int VPT = 8;  // 2048 / 256
  const int row = blockIdx.x;
  float* wr = W + (size_t)row * LY;
  _Float16* pr = P16 + (size_t)row * LY;
  const int tid = threadIdx.x, lane = tid & 31, wave = tid >> 5;

  float vals[VPT];
  float mx = -3.0e38f;
#pragma unroll
  for (int i = 0; i < VPT; ++i) {
    vals[i] = wr[tid + i * 256];
    mx = fmaxf(mx, vals[i]);
  }
#pragma unroll
  for (int off = 16; off > 0; off >>= 1) mx = fmaxf(mx, __shfl_xor(mx, off, 32));

  __shared__ float red[8];
  if (lane == 0) red[wave] = mx;
  __syncthreads();
  mx = red[0];
#pragma unroll
  for (int i = 1; i < 8; ++i) mx = fmaxf(mx, red[i]);

  float s = 0.0f;
#pragma unroll
  for (int i = 0; i < VPT; ++i) {
    const float e = __expf(vals[i] - mx);
    vals[i] = e;
    s += e;
  }
#pragma unroll
  for (int off = 16; off > 0; off >>= 1) s += __shfl_xor(s, off, 32);
  __syncthreads();
  if (lane == 0) red[wave] = s;
  __syncthreads();
  s = 0.0f;
#pragma unroll
  for (int i = 0; i < 8; ++i) s += red[i];
  const float inv = 1.0f / s;

#pragma unroll
  for (int i = 0; i < VPT; ++i) {
    const float p = vals[i] * inv;
    wr[tid + i * 256] = p;
    pr[tid + i * 256] = (_Float16)p;
  }
}

// ===========================================================================
extern "C" void kernel_launch(void* const* d_in, const int* in_sizes, int n_in,
                              void* d_out, int out_size, void* d_ws,
                              size_t ws_size, hipStream_t stream) {
  constexpr int Bn = 16, LX = 2048, LY = 2048, XS = 1024, AT = 1024;

  const float* x  = (const float*)d_in[0];
  const float* y  = (const float*)d_in[1];
  const float* wq = (const float*)d_in[2];
  const float* bq = (const float*)d_in[3];
  const float* wk = (const float*)d_in[4];
  const float* bk = (const float*)d_in[5];
  const float* wv = (const float*)d_in[6];
  const float* bv = (const float*)d_in[7];
  const float* wg = (const float*)d_in[8];
  const unsigned char* masks = (const unsigned char*)d_in[9];  // jnp bool

  // Workspace layout (328 MiB, stream-ordered overlays):
  //   [0,8M)     : wqT|wkT|wvT|wgT  (f16 transposed weights, 2 MiB each)
  //   [8M,72M)   : x16  -> kpre (after q GEMM) -> low half of P16
  //   [72M,136M) : y16  -> high half of P16 (after v/k GEMMs)
  //   [136M,200M): q16   [200M,264M): k16   [264M,328M): vT16
  char* ws = (char*)d_ws;
  const size_t MiB = 1ull << 20;
  _Float16* wqT = (_Float16*)(ws + 0 * MiB);
  _Float16* wkT = (_Float16*)(ws + 2 * MiB);
  _Float16* wvT = (_Float16*)(ws + 4 * MiB);
  _Float16* wgT = (_Float16*)(ws + 6 * MiB);
  _Float16* x16 = (_Float16*)(ws + 8 * MiB);
  _Float16* y16 = (_Float16*)(ws + 72 * MiB);
  _Float16* q16 = (_Float16*)(ws + 136 * MiB);
  _Float16* k16 = (_Float16*)(ws + 200 * MiB);
  _Float16* vT16 = (_Float16*)(ws + 264 * MiB);  // per-batch [AT, LY]
  _Float16* kpre = x16;                          // x16 dead after q GEMM
  _Float16* P16 = (_Float16*)(ws + 8 * MiB);     // x16+y16 dead after k GEMM

  float* attnOut = (float*)d_out;                   // [B, LX, AT] f32
  float* wOut = attnOut + (size_t)Bn * LX * AT;     // [B, LX, LY] f32

  const long long nx = (long long)Bn * LX * XS;     // 33.5M

  // 1) conversions: activations straight, weights transposed (N-major)
  cvt_f32_f16<<<dim3((unsigned)(nx / 4 / 256)), 256, 0, stream>>>(x, x16, nx);
  cvt_f32_f16<<<dim3((unsigned)(nx / 4 / 256)), 256, 0, stream>>>(y, y16, nx);
  const dim3 gT(XS / 32, AT / 32);
  cvt_transpose_f32_f16<<<gT, 256, 0, stream>>>(wq, wqT, XS, AT);
  cvt_transpose_f32_f16<<<gT, 256, 0, stream>>>(wk, wkT, XS, AT);
  cvt_transpose_f32_f16<<<gT, 256, 0, stream>>>(wv, wvT, XS, AT);
  cvt_transpose_f32_f16<<<gT, 256, 0, stream>>>(wg, wgT, AT, AT);

  // 2) projections (f32 accum, f16 out)
  const dim3 gProj(AT / BN, (Bn * LX) / BM, 1);
  gemm_nt<0, 1, 1, 0><<<gProj, 256, 0, stream>>>(      // q = relu(x@wq+bq)
      x16, wqT, bq, nullptr, q16, Bn * LX, AT, XS, 0.f, 0, 0, 0, 0);
  gemm_nt<0, 1, 1, 0><<<gProj, 256, 0, stream>>>(      // kpre = relu(y@wk+bk)
      y16, wkT, bk, nullptr, kpre, Bn * LY, AT, XS, 0.f, 0, 0, 0, 0);
  gemm_nt<2, 1, 1, 0><<<dim3(AT / BN, LY / BM, Bn), 256, 0, stream>>>(
      // vT[b][a][ly] = relu(y@wv+bv): transposed store feeds P@v as N-major B
      y16, wvT, bv, nullptr, vT16, LY, AT, XS, 0.f,
      (unsigned)(LY * XS), 0, (unsigned)(AT * LY), 0);
  gemm_nt<0, 0, 0, 0><<<gProj, 256, 0, stream>>>(      // k = kpre @ wg
      kpre, wgT, nullptr, nullptr, k16, Bn * LY, AT, AT, 0.f, 0, 0, 0, 0);

  // 3) masked scaled scores (NT: B = k16, N-major already) -> w region (raw)
  gemm_nt<1, 0, 0, 1><<<dim3(LY / BN, LX / BM, Bn), 256, 0, stream>>>(
      q16, k16, nullptr, masks, wOut, LX, LY, AT, 0.03125f /* 1/sqrt(1024) */,
      (unsigned)(LX * AT), (unsigned)(LY * AT), (unsigned)(LX * LY),
      (unsigned)LY);

  // 4) softmax rows: normalize f32 in place + emit f16 probabilities
  softmax_rows<<<dim3(Bn * LX), 256, 0, stream>>>(wOut, P16, LY);

  // 5) attention = P @ v  (NT with Bt = vT16), f32 out
  gemm_nt<1, 0, 0, 0><<<dim3(AT / BN, LX / BM, Bn), 256, 0, stream>>>(
      P16, vT16, nullptr, nullptr, attnOut, LX, AT, LY, 0.f,
      (unsigned)(LX * LY), (unsigned)(AT * LY), (unsigned)(LX * AT), 0);
}
